// single_task_model_70351564308756
// MI455X (gfx1250) — matrix operand order repaired and checked by reading, not compiled, not run on hardware
//
#include <hip/hip_runtime.h>

typedef __attribute__((ext_vector_type(16))) __bf16 v16bf;
typedef __attribute__((ext_vector_type(8)))  __bf16 v8bf;
typedef __attribute__((ext_vector_type(8)))  float  v8f;

#define NN 100000
#define EE 400000
#define GG 4096
#define DD 300
#define HH 600
#define LL 5
#define BN_EPS 1e-5f
#define SBK 616            // padded row stride (bf16 elems) of the B panel in LDS
#define ATILE (128 * 32)   // A tile elems per buffer

// ---------------------------------------------------------------------------
// Node input embedding: h[n,d] = atom_emb1[x[n,0],d] + atom_emb2[x[n,1],d]
// one thread per float4 (D = 300 = 75 float4)
// ---------------------------------------------------------------------------
__global__ void embed_kernel(const int* __restrict__ x,
                             const float* __restrict__ ae1,
                             const float* __restrict__ ae2,
                             float* __restrict__ h, int totalq) {
  int idx = blockIdx.x * blockDim.x + threadIdx.x;
  if (idx >= totalq) return;
  int n = idx / 75;
  int q = (idx - n * 75) * 4;
  int t0 = x[n * 2 + 0];
  int t1 = x[n * 2 + 1];
  float4 a = *(const float4*)(ae1 + (size_t)t0 * DD + q);
  float4 b = *(const float4*)(ae2 + (size_t)t1 * DD + q);
  float4 r = {a.x + b.x, a.y + b.y, a.z + b.z, a.w + b.w};
  *(float4*)(h + (size_t)n * DD + q) = r;
}

// ---------------------------------------------------------------------------
// agg[n,d] = h[n,d] + self_emb[d]   (self loop: ee1 row 4 + ee2 row 0)
// ---------------------------------------------------------------------------
__global__ void agg_init_kernel(const float* __restrict__ h,
                                const float* __restrict__ ee1l,
                                const float* __restrict__ ee2l,
                                float* __restrict__ agg, int totalq) {
  int idx = blockIdx.x * blockDim.x + threadIdx.x;
  if (idx >= totalq) return;
  int n = idx / 75;
  int q = (idx - n * 75) * 4;
  float4 hv = *(const float4*)(h + (size_t)n * DD + q);
  float4 s1 = *(const float4*)(ee1l + 4 * DD + q);
  float4 s2 = *(const float4*)(ee2l + q);
  float4 r = {hv.x + s1.x + s2.x, hv.y + s1.y + s2.y,
              hv.z + s1.z + s2.z, hv.w + s1.w + s2.w};
  *(float4*)(agg + (size_t)n * DD + q) = r;
}

// ---------------------------------------------------------------------------
// Edge message scatter: agg[dst] += h[src] + edge_emb  (one block per edge,
// 96 threads, thread q handles float4 chunk q)
// ---------------------------------------------------------------------------
__global__ void edge_scatter_kernel(const float* __restrict__ h,
                                    const int* __restrict__ src,
                                    const int* __restrict__ dst,
                                    const int* __restrict__ ea,
                                    const float* __restrict__ ee1l,
                                    const float* __restrict__ ee2l,
                                    float* __restrict__ agg) {
  int e = blockIdx.x;
  int q = threadIdx.x;
  if (q >= 75) return;
  int d = q * 4;
  int s  = src[e];
  int t  = dst[e];
  int a0 = ea[e * 2 + 0];
  int a1 = ea[e * 2 + 1];
  float4 hv = *(const float4*)(h    + (size_t)s  * DD + d);
  float4 v1 = *(const float4*)(ee1l + (size_t)a0 * DD + d);
  float4 v2 = *(const float4*)(ee2l + (size_t)a1 * DD + d);
  float* p = agg + (size_t)t * DD + d;
  atomicAdd(p + 0, hv.x + v1.x + v2.x);
  atomicAdd(p + 1, hv.y + v1.y + v2.y);
  atomicAdd(p + 2, hv.z + v1.z + v2.z);
  atomicAdd(p + 3, hv.w + v1.w + v2.w);
}

// ---------------------------------------------------------------------------
// WMMA bf16 GEMM:  C[M,Nc] = epilogue(A[M,K] @ B[K,Nc] + bias)
// 256 thr = 8 waves (4x2); block tile 128x64; K-step 32.
// B panel (64 x Kpad, bf16) staged ONCE into LDS; A tiles double-buffered.
// All staging is branch-free: addresses are clamped in-bounds and K-tail
// elements are zeroed with selects (out-of-range M/N rows are masked at the
// epilogue store, so clamped garbage never reaches valid outputs).
// mode 0: relu(acc+bias) | mode 1: bn(acc+bias) | mode 2: relu(bn(acc+bias))
// ---------------------------------------------------------------------------
__device__ inline v16bf load_frag16(const __bf16* __restrict__ p, int gap) {
  v16bf v;
#pragma unroll
  for (int i = 0; i < 8; ++i) {
    v[i]     = p[i];
    v[8 + i] = p[gap + i];
  }
  return v;
}

__device__ inline float4 zero_if(float4 f, bool z) {
  float4 r;
  r.x = z ? 0.f : f.x;
  r.y = z ? 0.f : f.y;
  r.z = z ? 0.f : f.z;
  r.w = z ? 0.f : f.w;
  return r;
}

__global__ __launch_bounds__(256)
void gemm_bf16_wmma(const float* __restrict__ A, const float* __restrict__ B,
                    const float* __restrict__ bias, float* __restrict__ C,
                    int M, int Nc, int K,
                    const float* __restrict__ bng, const float* __restrict__ bnb,
                    const float* __restrict__ bnm, const float* __restrict__ bnv,
                    int mode) {
  extern __shared__ __bf16 smem[];
  __bf16* sB = smem;                 // [64][SBK]
  __bf16* sA = smem + 64 * SBK;      // [2][128*32]

  const int bM  = blockIdx.y * 128;
  const int bN  = blockIdx.x * 64;
  const int tid = threadIdx.x;
  const int Kc32 = (K + 31) & ~31;

  // ---- stage entire B panel (bf16, [n][k], zero-padded) once ----
  {
    const int kq = tid >> 4;          // 0..15
    const int n4 = (tid & 15) * 4;    // 0..60
    int gn = bN + n4;
    if (gn > Nc - 4) gn = Nc - 4;     // clamp; garbage cols masked at store
    for (int k = kq; k < Kc32; k += 16) {
      int kc = (k < K) ? k : (K - 1); // clamp row
      float4 f = *(const float4*)(B + (size_t)kc * Nc + gn);
      f = zero_if(f, k >= K);         // exact zeros in K padding
      sB[(n4 + 0) * SBK + k] = (__bf16)f.x;
      sB[(n4 + 1) * SBK + k] = (__bf16)f.y;
      sB[(n4 + 2) * SBK + k] = (__bf16)f.z;
      sB[(n4 + 3) * SBK + k] = (__bf16)f.w;
    }
  }

  // ---- A tile staging: 16 elems/thread as two 8-wide bf16 vector stores ----
  const int arow = tid >> 1;               // 0..127
  const int akb  = (tid & 1) * 16;         // 0 or 16
  int crow = bM + arow;
  if (crow > M - 1) crow = M - 1;          // clamp; garbage rows masked at store
  const float* Ar = A + (size_t)crow * K;

  auto stageA = [&](int bufi, int k0) {
#pragma unroll
    for (int c = 0; c < 2; ++c) {
      int k   = akb + c * 8;
      int gk  = k0 + k;
      int g0  = (gk     < K - 4) ? gk       : (K - 4);  // K % 4 == 0
      int g1  = (gk + 4 < K - 4) ? (gk + 4) : (K - 4);
      float4 f0 = *(const float4*)(Ar + g0);
      float4 f1 = *(const float4*)(Ar + g1);
      f0 = zero_if(f0, gk >= K);
      f1 = zero_if(f1, gk + 4 >= K);
      v8bf pk;
      pk[0] = (__bf16)f0.x; pk[1] = (__bf16)f0.y;
      pk[2] = (__bf16)f0.z; pk[3] = (__bf16)f0.w;
      pk[4] = (__bf16)f1.x; pk[5] = (__bf16)f1.y;
      pk[6] = (__bf16)f1.z; pk[7] = (__bf16)f1.w;
      *(v8bf*)&sA[bufi * ATILE + arow * 32 + k] = pk;
    }
    __builtin_prefetch(Ar + k0 + 64, 0, 0);   // global_prefetch next tiles
  };

  stageA(0, 0);
  __syncthreads();

  const int lane = tid & 31;
  const int w    = tid >> 5;
  const int wm   = (w & 3) * 32;
  const int wn   = (w >> 2) * 32;
  const int r    = lane & 15;
  const int kh   = lane >> 4;

  v8f acc00 = {0,0,0,0,0,0,0,0};
  v8f acc01 = acc00, acc10 = acc00, acc11 = acc00;

  int buf = 0;
  for (int k0 = 0; k0 < K; k0 += 32) {
    if (k0 + 32 < K) stageA(buf ^ 1, k0 + 32);   // overlap with WMMA below

    const __bf16* a0p = &sA[buf * ATILE + (wm + r)      * 32 + kh * 8];
    const __bf16* a1p = &sA[buf * ATILE + (wm + 16 + r) * 32 + kh * 8];
    const __bf16* b0p = &sB[(wn + r)      * SBK + k0 + kh * 8];
    const __bf16* b1p = &sB[(wn + 16 + r) * SBK + k0 + kh * 8];
    v16bf a0 = load_frag16(a0p, 16);
    v16bf a1 = load_frag16(a1p, 16);
    v16bf b0 = load_frag16(b0p, 16);
    v16bf b1 = load_frag16(b1p, 16);

    acc00 = __builtin_amdgcn_wmma_f32_16x16x32_bf16(false, a0, false, b0, (short)0, acc00, false, false);
    acc01 = __builtin_amdgcn_wmma_f32_16x16x32_bf16(false, a0, false, b1, (short)0, acc01, false, false);
    acc10 = __builtin_amdgcn_wmma_f32_16x16x32_bf16(false, a1, false, b0, (short)0, acc10, false, false);
    acc11 = __builtin_amdgcn_wmma_f32_16x16x32_bf16(false, a1, false, b1, (short)0, acc11, false, false);

    __syncthreads();
    buf ^= 1;
  }

  // epilogue: C/D layout -> VGPR i : M = i + 8*(lane>=16), N = lane&15
  const int kh8 = kh * 8;
#pragma unroll
  for (int ti = 0; ti < 2; ++ti) {
#pragma unroll
    for (int tj = 0; tj < 2; ++tj) {
      const v8f acc = (ti == 0) ? (tj == 0 ? acc00 : acc01)
                                : (tj == 0 ? acc10 : acc11);
      int gn = bN + wn + tj * 16 + r;
      if (gn >= Nc) continue;
      float bv = bias[gn];
      float g = 1.f, bb = 0.f, mn = 0.f, inv = 1.f;
      if (mode != 0) {
        g   = bng[gn];
        bb  = bnb[gn];
        mn  = bnm[gn];
        inv = rsqrtf(bnv[gn] + BN_EPS);
      }
#pragma unroll
      for (int i = 0; i < 8; ++i) {
        int gm = bM + wm + ti * 16 + i + kh8;
        if (gm >= M) continue;
        float v = acc[i] + bv;
        if (mode == 0) {
          v = fmaxf(v, 0.0f);
        } else {
          v = (v - mn) * inv * g + bb;
          if (mode == 2) v = fmaxf(v, 0.0f);
        }
        C[(size_t)gm * Nc + gn] = v;
      }
    }
  }
}

// ---------------------------------------------------------------------------
// pooled zeroing + graph pooling scatter (float4 granularity)
// ---------------------------------------------------------------------------
__global__ void zero_kernel(float4* __restrict__ p, int totalq) {
  int idx = blockIdx.x * blockDim.x + threadIdx.x;
  if (idx < totalq) p[idx] = (float4){0.f, 0.f, 0.f, 0.f};
}

__global__ void pool_scatter_kernel(const float* __restrict__ h,
                                    const int* __restrict__ batch,
                                    float* __restrict__ pooled, int totalq) {
  int idx = blockIdx.x * blockDim.x + threadIdx.x;
  if (idx >= totalq) return;
  int n = idx / 75;
  int d = (idx - n * 75) * 4;
  int g = batch[n];
  float4 v = *(const float4*)(h + (size_t)n * DD + d);
  float* p = pooled + (size_t)g * DD + d;
  atomicAdd(p + 0, v.x);
  atomicAdd(p + 1, v.y);
  atomicAdd(p + 2, v.z);
  atomicAdd(p + 3, v.w);
}

// ---------------------------------------------------------------------------
// Final head layer: logits[g,0:2] = hidden[g,:] @ head_w2 + head_b2
// ---------------------------------------------------------------------------
__global__ void logits_kernel(const float* __restrict__ hidden,
                              const float* __restrict__ w2,
                              const float* __restrict__ b2,
                              float* __restrict__ out, int G) {
  int g = blockIdx.x * blockDim.x + threadIdx.x;
  if (g >= G) return;
  float a0 = b2[0], a1 = b2[1];
  const float4* hr = (const float4*)(hidden + (size_t)g * 512);
#pragma unroll 4
  for (int j4 = 0; j4 < 128; ++j4) {
    float4 v = hr[j4];
    int j = j4 * 4;
    a0 += v.x * w2[(j + 0) * 2] + v.y * w2[(j + 1) * 2] +
          v.z * w2[(j + 2) * 2] + v.w * w2[(j + 3) * 2];
    a1 += v.x * w2[(j + 0) * 2 + 1] + v.y * w2[(j + 1) * 2 + 1] +
          v.z * w2[(j + 2) * 2 + 1] + v.w * w2[(j + 3) * 2 + 1];
  }
  out[g * 2 + 0] = a0;
  out[g * 2 + 1] = a1;
}

// ---------------------------------------------------------------------------
extern "C" void kernel_launch(void* const* d_in, const int* in_sizes, int n_in,
                              void* d_out, int out_size, void* d_ws, size_t ws_size,
                              hipStream_t stream) {
  const int*   x          = (const int*)d_in[0];
  const int*   edge_index = (const int*)d_in[1];   // [2,E]: src then dst
  const int*   edge_attr  = (const int*)d_in[2];   // [E,2]
  const int*   batch      = (const int*)d_in[3];
  const float* atom_emb1  = (const float*)d_in[4];
  const float* atom_emb2  = (const float*)d_in[5];
  const float* edge_emb1  = (const float*)d_in[6]; // [L,6,D]
  const float* edge_emb2  = (const float*)d_in[7]; // [L,3,D]
  const float* w1         = (const float*)d_in[8];  // [L,D,H]
  const float* b1         = (const float*)d_in[9];  // [L,H]
  const float* w2         = (const float*)d_in[10]; // [L,H,D]
  const float* b2         = (const float*)d_in[11]; // [L,D]
  const float* bn_g       = (const float*)d_in[12];
  const float* bn_b       = (const float*)d_in[13];
  const float* bn_m       = (const float*)d_in[14];
  const float* bn_v       = (const float*)d_in[15];
  const float* head_w1    = (const float*)d_in[16]; // [D,512]
  const float* head_b1    = (const float*)d_in[17];
  const float* head_w2    = (const float*)d_in[18]; // [512,2]
  const float* head_b2    = (const float*)d_in[19];

  float* ws = (float*)d_ws;
  float* h      = ws;                                         // N*D
  float* agg    = ws + (size_t)NN * DD;                       // N*D
  float* mid    = ws + 2 * (size_t)NN * DD;                   // N*H
  float* pooled = ws + 2 * (size_t)NN * DD + (size_t)NN * HH; // G*D
  float* hidden = pooled + (size_t)GG * DD;                   // G*512

  const int tpb   = 256;
  const int totNQ = NN * 75;          // N*D in float4 units
  const size_t gemm_lds = (size_t)(64 * SBK + 2 * ATILE) * sizeof(__bf16);

  // node embedding
  embed_kernel<<<(totNQ + tpb - 1) / tpb, tpb, 0, stream>>>(x, atom_emb1, atom_emb2, h, totNQ);

  for (int l = 0; l < LL; ++l) {
    const float* ee1l = edge_emb1 + (size_t)l * 6 * DD;
    const float* ee2l = edge_emb2 + (size_t)l * 3 * DD;

    // agg = h + self_emb
    agg_init_kernel<<<(totNQ + tpb - 1) / tpb, tpb, 0, stream>>>(h, ee1l, ee2l, agg, totNQ);

    // agg[dst] += h[src] + edge_emb
    edge_scatter_kernel<<<EE, 96, 0, stream>>>(h, edge_index, edge_index + EE,
                                               edge_attr, ee1l, ee2l, agg);

    // mid = relu(agg @ w1[l] + b1[l])            [N,D]x[D,H]
    dim3 g1((HH + 63) / 64, (NN + 127) / 128);
    gemm_bf16_wmma<<<g1, 256, gemm_lds, stream>>>(agg, w1 + (size_t)l * DD * HH,
                                                  b1 + (size_t)l * HH, mid,
                                                  NN, HH, DD,
                                                  nullptr, nullptr, nullptr, nullptr, 0);

    // h = bn(mid @ w2[l] + b2[l]) [+relu]        [N,H]x[H,D]
    dim3 g2((DD + 63) / 64, (NN + 127) / 128);
    gemm_bf16_wmma<<<g2, 256, gemm_lds, stream>>>(mid, w2 + (size_t)l * HH * DD,
                                                  b2 + (size_t)l * DD, h,
                                                  NN, DD, HH,
                                                  bn_g + (size_t)l * DD, bn_b + (size_t)l * DD,
                                                  bn_m + (size_t)l * DD, bn_v + (size_t)l * DD,
                                                  (l < LL - 1) ? 2 : 1);
  }

  // graph pooling
  const int totGQ = GG * DD / 4;
  zero_kernel<<<(totGQ + tpb - 1) / tpb, tpb, 0, stream>>>((float4*)pooled, totGQ);
  pool_scatter_kernel<<<(totNQ + tpb - 1) / tpb, tpb, 0, stream>>>(h, batch, pooled, totNQ);

  // head layer 1: hidden = relu(pooled @ head_w1 + head_b1)   [G,D]x[D,512]
  dim3 gh((512 + 63) / 64, (GG + 127) / 128);
  gemm_bf16_wmma<<<gh, 256, gemm_lds, stream>>>(pooled, head_w1, head_b1, hidden,
                                                GG, 512, DD,
                                                nullptr, nullptr, nullptr, nullptr, 0);

  // head layer 2: logits
  logits_kernel<<<(GG + tpb - 1) / tpb, tpb, 0, stream>>>(hidden, head_w2, head_b2,
                                                          (float*)d_out, GG);
}
